// TransformerAttentionSepHeToHoModule_9328668967248
// MI455X (gfx1250) — compile-verified, matrix operand order checked by reading, not run on hardware
//
#include <hip/hip_runtime.h>

#define NN   8192
#define DIM  256
#define H_   8
#define HD_  32
#define HID  2048

typedef __attribute__((ext_vector_type(16))) _Float16 v16h;
typedef __attribute__((ext_vector_type(8)))  _Float16 h8;
typedef __attribute__((ext_vector_type(8)))  float    f32x8;

// ---------------------------------------------------------------- WMMA core
__device__ __forceinline__ f32x8 wmma16(v16h a, v16h b, f32x8 c) {
  // D = A(16x32 f16) * B(32x16 f16) + C(16x16 f32)
  return __builtin_amdgcn_wmma_f32_16x16x32_f16(
      /*neg_a=*/false, a, /*neg_b=*/false, b,
      /*c_mod=*/(short)0, c, /*reuse_a=*/false, /*reuse_b=*/false);
}

// Load one 16x32 f16 fragment (A-layout; B uses the same lane pattern with
// row = N). Lane L (0-15): M=L, K = {0..7, 16..23}; lane L+16: K = {8..15, 24..31}.
__device__ __forceinline__ v16h load_frag(const _Float16* base, int ld,
                                          int row0, int k0, int lane) {
  int m  = lane & 15;
  int kh = (lane >> 4) & 1;
  const _Float16* p = base + (size_t)(row0 + m) * ld + k0 + kh * 8;
  h8 lo = *(const h8*)p;
  h8 hi = *(const h8*)(p + 16);
  v16h r;
#pragma unroll
  for (int i = 0; i < 8; ++i) { r[i] = lo[i]; r[i + 8] = hi[i]; }
  return r;
}

// ---------------------------------------------------------------- utilities
__global__ void k_zero_f32(float* p, long n) {
  long i = (long)blockIdx.x * blockDim.x + threadIdx.x;
  long s = (long)gridDim.x * blockDim.x;
  for (; i < n; i += s) p[i] = 0.f;
}
__global__ void k_zero_i32(int* p, long n) {
  long i = (long)blockIdx.x * blockDim.x + threadIdx.x;
  long s = (long)gridDim.x * blockDim.x;
  for (; i < n; i += s) p[i] = 0;
}
__global__ void k_i2f(const int* a, float* b, long n) {
  long i = (long)blockIdx.x * blockDim.x + threadIdx.x;
  if (i < n) b[i] = (float)a[i];
}
__global__ void k_f2h(const float* a, _Float16* b, long n) {
  long i = (long)blockIdx.x * blockDim.x + threadIdx.x;
  if (i < n) b[i] = (_Float16)a[i];
}

// ---------------------------------------------------------------- graph part
__global__ void k_degrees(const int* __restrict__ src, const int* __restrict__ dst,
                          int* degI, int* inI, int E) {
  int e = blockIdx.x * blockDim.x + threadIdx.x;
  if (e < E) {
    atomicAdd(&degI[src[e]], 1);
    atomicAdd(&inI[dst[e]], 1);
  }
}

// one wave per edge; 4 scatter-accumulated messages per dim
__global__ void k_edge_agg(const float* __restrict__ x, const int* __restrict__ src,
                           const int* __restrict__ dst, const float* __restrict__ degF,
                           float* __restrict__ acc, int E) {
  int lane = threadIdx.x & 31;
  int e = blockIdx.x * (blockDim.x >> 5) + (threadIdx.x >> 5);
  if (e >= E) return;
  int s = src[e], t = dst[e];
  float c  = rsqrtf(degF[s] * degF[t]);
  float ic = 1.f / c;
  const float* xr = x + (size_t)s * DIM;
  float* a0 = acc + (size_t)t * DIM;
  const size_t P = (size_t)NN * DIM;
  for (int d = lane; d < DIM; d += 32) {
    float xs = xr[d];
    atomicAdd(a0 + d,         xs + c);
    atomicAdd(a0 + d + P,     xs - c);
    atomicAdd(a0 + d + 2 * P, xs * c);
    atomicAdd(a0 + d + 3 * P, xs * ic);
  }
}

__global__ void k_emean_div(float* acc, const float* __restrict__ inF) {
  size_t idx = (size_t)blockIdx.x * blockDim.x + threadIdx.x;
  const size_t P = (size_t)NN * DIM;
  if (idx >= P) return;
  float inv = 1.f / inF[idx / DIM];
#pragma unroll
  for (int w = 0; w < 4; ++w) acc[w * P + idx] *= inv;
}

// x_concat = (x1, x2, x0, x, x3) -> f16, N x 1280
__global__ void k_build_xc(const float* __restrict__ x, const float* __restrict__ acc,
                           _Float16* __restrict__ xc) {
  size_t idx = (size_t)blockIdx.x * blockDim.x + threadIdx.x;
  const size_t total = (size_t)NN * 1280;
  if (idx >= total) return;
  int n = (int)(idx / 1280), col = (int)(idx % 1280);
  int sel = col >> 8, d = col & 255;
  const size_t P = (size_t)NN * DIM;
  size_t nd = (size_t)n * DIM + d;
  float v;
  if      (sel == 0) v = acc[P + nd];      // x1 = mean(xs - c)
  else if (sel == 1) v = acc[2 * P + nd];  // x2 = mean(xs * c)
  else if (sel == 2) v = acc[nd];          // x0 = mean(xs + c)
  else if (sel == 3) v = x[nd];            // x
  else               v = acc[3 * P + nd];  // x3 = mean(xs / c)
  xc[idx] = (_Float16)v;
}

// ---------------------------------------------------------------- WMMA GEMM
// C[m,n] = sum_k A[m,k] * B[n,k] + bias[n]   (A: MxK f16, B: NxK f16)
__global__ void k_gemm(const _Float16* __restrict__ A, const _Float16* __restrict__ B,
                       const float* __restrict__ bias, float* Cf, _Float16* Ch,
                       int M, int Nn, int K, int doGelu) {
  int lane = threadIdx.x & 31;
  int nT = Nn >> 4;
  long tile = (long)blockIdx.x * (blockDim.x >> 5) + (threadIdx.x >> 5);
  long totalTiles = (long)(M >> 4) * nT;
  if (tile >= totalTiles) return;
  int m0 = (int)(tile / nT) << 4;
  int n0 = (int)(tile % nT) << 4;
  int m = lane & 15, kh = (lane >> 4) & 1;
  const _Float16* pa = A + (size_t)(m0 + m) * K + kh * 8;
  const _Float16* pb = B + (size_t)(n0 + m) * K + kh * 8;
  f32x8 acc = {};
  for (int k0 = 0; k0 < K; k0 += 32) {
    __builtin_prefetch(pa + k0 + 128, 0, 0);  // global_prefetch_b8
    v16h a = load_frag(A, K, m0, k0, lane);
    v16h b = load_frag(B, K, n0, k0, lane);
    acc = wmma16(a, b, acc);
  }
  (void)pa; (void)pb;
  int col = n0 + (lane & 15);
  int rbase = m0 + ((lane >> 4) << 3);
  float bv = bias ? bias[col] : 0.f;
#pragma unroll
  for (int i = 0; i < 8; ++i) {
    float xv = acc[i] + bv;
    if (doGelu) xv = 0.5f * xv * (1.f + erff(xv * 0.70710678118f));
    size_t idx = (size_t)(rbase + i) * Nn + col;
    if (Cf) Cf[idx] = xv;
    if (Ch) Ch[idx] = (_Float16)xv;
  }
}

// ------------------------------------------------------- global stats (q,k,v)
__global__ void k_reduce_qkv(const float* __restrict__ q, const float* __restrict__ k,
                             const float* __restrict__ v, double* __restrict__ stats) {
  const size_t total = (size_t)NN * DIM;
  size_t i = (size_t)blockIdx.x * blockDim.x + threadIdx.x;
  size_t stride = (size_t)gridDim.x * blockDim.x;
  float s[6] = {0, 0, 0, 0, 0, 0};
  for (; i < total; i += stride) {
    float a = q[i], b = k[i], c = v[i];
    s[0] += a; s[1] += a * a; s[2] += b; s[3] += b * b; s[4] += c; s[5] += c * c;
  }
#pragma unroll
  for (int off = 16; off >= 1; off >>= 1)
#pragma unroll
    for (int j = 0; j < 6; ++j) s[j] += __shfl_down(s[j], off, 32);
  if ((threadIdx.x & 31) == 0)
#pragma unroll
    for (int j = 0; j < 6; ++j) atomicAdd(&stats[j], (double)s[j]);
}

__global__ void k_normalize(float* q, float* k, const float* __restrict__ v,
                            const double* __restrict__ stats, _Float16* __restrict__ vT) {
  size_t idx = (size_t)blockIdx.x * blockDim.x + threadIdx.x;
  const size_t total = (size_t)NN * DIM;
  if (idx >= total) return;
  const double M = (double)total;
  float mq = (float)(stats[0] / M);
  float sq = (float)sqrt((stats[1] - stats[0] * stats[0] / M) / (M - 1.0));
  float mk = (float)(stats[2] / M);
  float sk = (float)sqrt((stats[3] - stats[2] * stats[2] / M) / (M - 1.0));
  float sv = (float)sqrt((stats[5] - stats[4] * stats[4] / M) / (M - 1.0));
  int i = (int)(idx / DIM), d = (int)(idx % DIM);
  vT[(size_t)d * NN + i] = (_Float16)((v[idx] - mq) / sv);  // transposed f16 V
  q[idx] = (q[idx] - mq) / sq;
  k[idx] = (k[idx] - mk) / sk;
}

// per-node head-softmax score + global sum/sumsq of score
__global__ void k_score(const float* __restrict__ q, const float* __restrict__ k,
                        float* __restrict__ score, double* __restrict__ stats) {
  int n = blockIdx.x * blockDim.x + threadIdx.x;
  float s[H_];
  if (n < NN) {
    const float* qr = q + (size_t)n * DIM;
    const float* kr = k + (size_t)n * DIM;
    float mx = -3.0e38f;
#pragma unroll
    for (int h = 0; h < H_; ++h) {
      float d = 0.f;
      for (int j = 0; j < HD_; ++j) d += qr[h * HD_ + j] * kr[h * HD_ + j];
      s[h] = d; mx = fmaxf(mx, d);
    }
    float z = 0.f;
#pragma unroll
    for (int h = 0; h < H_; ++h) { s[h] = __expf(s[h] - mx); z += s[h]; }
    float inv = 1.f / z;
#pragma unroll
    for (int h = 0; h < H_; ++h) { s[h] *= inv; score[(size_t)n * H_ + h] = s[h]; }
  } else {
#pragma unroll
    for (int h = 0; h < H_; ++h) s[h] = 0.f;
  }
  float ps = 0.f, pss = 0.f;
#pragma unroll
  for (int h = 0; h < H_; ++h) { ps += s[h]; pss += s[h] * s[h]; }
#pragma unroll
  for (int off = 16; off >= 1; off >>= 1) {
    ps += __shfl_down(ps, off, 32);
    pss += __shfl_down(pss, off, 32);
  }
  if ((threadIdx.x & 31) == 0) {
    atomicAdd(&stats[6], (double)ps);
    atomicAdd(&stats[7], (double)pss);
  }
}

// sim = (score-mean)/std + 0.004*noise, zero-padded K=8 -> 32 (f16, N x 32)
__global__ void k_sim(const float* __restrict__ score, const float* __restrict__ noise,
                      const double* __restrict__ stats, _Float16* __restrict__ simp) {
  size_t idx = (size_t)blockIdx.x * blockDim.x + threadIdx.x;
  const size_t total = (size_t)NN * 32;
  if (idx >= total) return;
  int n = (int)(idx >> 5), c = (int)(idx & 31);
  const double M = (double)NN * H_;
  float ms = (float)(stats[6] / M);
  float ss = (float)sqrt((stats[7] - stats[6] * stats[6] / M) / (M - 1.0));
  float v = 0.f;
  if (c < H_) {
    size_t j = (size_t)n * H_ + c;
    v = (score[j] - ms) / ss + 0.004f * noise[j];
  }
  simp[idx] = (_Float16)v;
}

// ----------------- pass 1: streaming row max / Z of S = sim @ sim^T via WMMA
__global__ void k_rowstats(const _Float16* __restrict__ simp,
                           float* __restrict__ rowM, float* __restrict__ rowZ) {
  int lane = threadIdx.x & 31;
  int t = blockIdx.x * (blockDim.x >> 5) + (threadIdx.x >> 5);
  int i0 = t << 4;
  if (i0 >= NN) return;
  v16h a = load_frag(simp, 32, i0, 0, lane);
  float m[8], z[8];
#pragma unroll
  for (int i = 0; i < 8; ++i) { m[i] = -3.0e38f; z[i] = 0.f; }
  for (int j0 = 0; j0 < NN; j0 += 16) {
    v16h b = load_frag(simp, 32, j0, 0, lane);
    f32x8 s = {};
    s = wmma16(a, b, s);
#pragma unroll
    for (int i = 0; i < 8; ++i) {
      float sv = s[i];
      float nm = fmaxf(m[i], sv);
      z[i] = z[i] * __expf(m[i] - nm) + __expf(sv - nm);
      m[i] = nm;
    }
  }
  // combine across the 16 lanes (columns) holding each row
#pragma unroll
  for (int mask = 8; mask >= 1; mask >>= 1)
#pragma unroll
    for (int i = 0; i < 8; ++i) {
      float om = __shfl_xor(m[i], mask, 32);
      float oz = __shfl_xor(z[i], mask, 32);
      float nm = fmaxf(m[i], om);
      z[i] = z[i] * __expf(m[i] - nm) + oz * __expf(om - nm);
      m[i] = nm;
    }
  if ((lane & 15) == 0) {
    int hi = lane >> 4;
#pragma unroll
    for (int i = 0; i < 8; ++i) {
      int r = i0 + i + (hi << 3);
      rowM[r] = m[i];
      rowZ[r] = z[i];
    }
  }
}

// ---- pass 2: out = P^T @ v, P regenerated 32x16 at a time through LDS + WMMA
__global__ void k_attn_out(const _Float16* __restrict__ simp, const float* __restrict__ rowM,
                           const float* __restrict__ rowZ, const _Float16* __restrict__ vT,
                           float* __restrict__ outF, _Float16* __restrict__ outH) {
  __shared__ __align__(16) _Float16 Pl[16 * 32];  // P^T tile: [j_local][i_local]
  int lane = threadIdx.x & 31;
  int wave = threadIdx.x >> 5;  // 8 waves
  int j0 = blockIdx.x << 4;
  int d0 = wave << 5;
  int col = lane & 15, hi = lane >> 4;
  v16h bj = load_frag(simp, 32, j0, 0, lane);  // fixed B fragment (rows j)
  f32x8 acc0 = {}, acc1 = {};
  for (int i0 = 0; i0 < NN; i0 += 32) {
    if (wave == 0) {
      v16h a0 = load_frag(simp, 32, i0, 0, lane);
      v16h a1 = load_frag(simp, 32, i0 + 16, 0, lane);
      f32x8 s0 = {}; s0 = wmma16(a0, bj, s0);
      f32x8 s1 = {}; s1 = wmma16(a1, bj, s1);
#pragma unroll
      for (int u = 0; u < 8; ++u) {
        int il = u + (hi << 3);
        int r0 = i0 + il;
        Pl[col * 32 + il] = (_Float16)(__expf(s0[u] - rowM[r0]) / rowZ[r0]);
        int r1 = i0 + 16 + il;
        Pl[col * 32 + 16 + il] = (_Float16)(__expf(s1[u] - rowM[r1]) / rowZ[r1]);
      }
    }
    __syncthreads();
    v16h ap = load_frag(Pl, 32, 0, 0, lane);           // P^T from LDS
    v16h b0 = load_frag(vT, NN, d0, i0, lane);         // V^T rows d0..d0+15
    v16h b1 = load_frag(vT, NN, d0 + 16, i0, lane);
    acc0 = wmma16(ap, b0, acc0);
    acc1 = wmma16(ap, b1, acc1);
    __syncthreads();
  }
#pragma unroll
  for (int u = 0; u < 8; ++u) {
    int r = j0 + u + (hi << 3);
    size_t ia = (size_t)r * DIM + d0 + col;
    outF[ia] = acc0[u];
    outH[ia] = (_Float16)acc0[u];
    size_t ib = (size_t)r * DIM + d0 + 16 + col;
    outF[ib] = acc1[u];
    outH[ib] = (_Float16)acc1[u];
  }
}

// ---------------------------------------------------------------- launcher
extern "C" void kernel_launch(void* const* d_in, const int* in_sizes, int n_in,
                              void* d_out, int out_size, void* d_ws, size_t ws_size,
                              hipStream_t stream) {
  const float* x    = (const float*)d_in[0];
  const int*   src  = (const int*)d_in[1];
  const int*   dst  = (const int*)d_in[2];
  const float* noise= (const float*)d_in[3];
  const float* Wc   = (const float*)d_in[4];
  const float* bc   = (const float*)d_in[5];
  const float* Wq   = (const float*)d_in[6];
  const float* bq   = (const float*)d_in[7];
  const float* Wk   = (const float*)d_in[8];
  const float* bk   = (const float*)d_in[9];
  const float* Wv   = (const float*)d_in[10];
  const float* bv   = (const float*)d_in[11];
  const float* W1   = (const float*)d_in[12];
  const float* b1   = (const float*)d_in[13];
  const float* W2   = (const float*)d_in[14];
  const float* b2   = (const float*)d_in[15];
  float* y = (float*)d_out;
  int E = in_sizes[1];

  char* p = (char*)d_ws;
  auto alloc = [&](size_t bytes) -> char* {
    char* r = p; p += (bytes + 255) & ~(size_t)255; return r;
  };
  const size_t P = (size_t)NN * DIM;
  int*       degI  = (int*)alloc(NN * 4);
  int*       inI   = (int*)alloc(NN * 4);
  float*     degF  = (float*)alloc(NN * 4);
  float*     inF   = (float*)alloc(NN * 4);
  float*     acc   = (float*)alloc(4 * P * 4);
  double*    stats = (double*)alloc(8 * 8);
  _Float16*  xc16  = (_Float16*)alloc((size_t)NN * 1280 * 2);
  _Float16*  Wc16  = (_Float16*)alloc((size_t)DIM * 1280 * 2);
  float*     xh    = (float*)alloc(P * 4);
  _Float16*  xh16  = (_Float16*)alloc(P * 2);
  _Float16*  Wq16  = (_Float16*)alloc((size_t)DIM * DIM * 2);
  _Float16*  Wk16  = (_Float16*)alloc((size_t)DIM * DIM * 2);
  _Float16*  Wv16  = (_Float16*)alloc((size_t)DIM * DIM * 2);
  float*     q     = (float*)alloc(P * 4);
  float*     k     = (float*)alloc(P * 4);
  float*     v     = (float*)alloc(P * 4);
  _Float16*  vT16  = (_Float16*)alloc(P * 2);
  float*     score = (float*)alloc((size_t)NN * H_ * 4);
  _Float16*  simp  = (_Float16*)alloc((size_t)NN * 32 * 2);
  float*     rowM  = (float*)alloc(NN * 4);
  float*     rowZ  = (float*)alloc(NN * 4);
  float*     attn  = (float*)alloc(P * 4);
  _Float16*  attn16= (_Float16*)alloc(P * 2);
  _Float16*  W116  = (_Float16*)alloc((size_t)HID * DIM * 2);
  _Float16*  W216  = (_Float16*)alloc((size_t)DIM * HID * 2);
  _Float16*  h16   = (_Float16*)alloc((size_t)NN * HID * 2);
  (void)ws_size; (void)n_in; (void)out_size;

  const int T = 256;
  auto nb = [](long n, int t) { return (unsigned)((n + t - 1) / t); };

  // --- graph aggregation ---
  k_zero_i32<<<nb(NN, T), T, 0, stream>>>(degI, NN);
  k_zero_i32<<<nb(NN, T), T, 0, stream>>>(inI, NN);
  k_zero_f32<<<2048, T, 0, stream>>>(acc, (long)4 * P);
  k_zero_f32<<<1, 32, 0, stream>>>((float*)stats, 16);
  k_degrees<<<nb(E, T), T, 0, stream>>>(src, dst, degI, inI, E);
  k_i2f<<<nb(NN, T), T, 0, stream>>>(degI, degF, NN);
  k_i2f<<<nb(NN, T), T, 0, stream>>>(inI, inF, NN);
  k_edge_agg<<<nb(E, 8), T, 0, stream>>>(x, src, dst, degF, acc, E);
  k_emean_div<<<nb(P, T), T, 0, stream>>>(acc, inF);
  k_build_xc<<<nb((long)NN * 1280, T), T, 0, stream>>>(x, acc, xc16);

  // --- weight conversions ---
  k_f2h<<<nb((long)DIM * 1280, T), T, 0, stream>>>(Wc, Wc16, (long)DIM * 1280);
  k_f2h<<<nb((long)DIM * DIM, T), T, 0, stream>>>(Wq, Wq16, (long)DIM * DIM);
  k_f2h<<<nb((long)DIM * DIM, T), T, 0, stream>>>(Wk, Wk16, (long)DIM * DIM);
  k_f2h<<<nb((long)DIM * DIM, T), T, 0, stream>>>(Wv, Wv16, (long)DIM * DIM);
  k_f2h<<<nb((long)HID * DIM, T), T, 0, stream>>>(W1, W116, (long)HID * DIM);
  k_f2h<<<nb((long)DIM * HID, T), T, 0, stream>>>(W2, W216, (long)DIM * HID);

  // --- xh = x_concat @ Wc^T + bc ; q/k/v ---
  k_gemm<<<(NN / 16) * (DIM / 16) / 8, T, 0, stream>>>(xc16, Wc16, bc, xh, xh16, NN, DIM, 1280, 0);
  k_gemm<<<(NN / 16) * (DIM / 16) / 8, T, 0, stream>>>(xh16, Wq16, bq, q, (_Float16*)nullptr, NN, DIM, DIM, 0);
  k_gemm<<<(NN / 16) * (DIM / 16) / 8, T, 0, stream>>>(xh16, Wk16, bk, k, (_Float16*)nullptr, NN, DIM, DIM, 0);
  k_gemm<<<(NN / 16) * (DIM / 16) / 8, T, 0, stream>>>(xh16, Wv16, bv, v, (_Float16*)nullptr, NN, DIM, DIM, 0);

  // --- normalizations, head softmax, sim ---
  k_reduce_qkv<<<2048, T, 0, stream>>>(q, k, v, stats);
  k_normalize<<<nb(P, T), T, 0, stream>>>(q, k, v, stats, vT16);
  k_score<<<nb(NN, T), T, 0, stream>>>(q, k, score, stats);
  k_sim<<<nb((long)NN * 32, T), T, 0, stream>>>(score, noise, stats, simp);

  // --- N x N attention, streamed (never materialized) ---
  k_rowstats<<<NN / 16 / 4, 128, 0, stream>>>(simp, rowM, rowZ);
  k_attn_out<<<NN / 16, T, 0, stream>>>(simp, rowM, rowZ, vT16, attn, attn16);

  // --- MLP ---
  k_gemm<<<(NN / 16) * (HID / 16) / 8, T, 0, stream>>>(attn16, W116, b1, (float*)nullptr, h16, NN, HID, DIM, 1);
  k_gemm<<<(NN / 16) * (DIM / 16) / 8, T, 0, stream>>>(h16, W216, b2, y, (_Float16*)nullptr, NN, DIM, HID, 0);
}
